// MLVAE_86199993631123
// MI455X (gfx1250) — matrix operand order repaired and checked by reading, not compiled, not run on hardware
//
#include <hip/hip_runtime.h>
#include <hip/hip_bf16.h>

typedef __attribute__((ext_vector_type(16))) _Float16 v16h;
typedef __attribute__((ext_vector_type(8)))  _Float16 v8h;
typedef __attribute__((ext_vector_type(8)))  float    v8f;
typedef __attribute__((ext_vector_type(4)))  int      v4i_;

#define N_B   64
#define L_IN  2048
#define C1_   64
#define C2_   128
#define C3_   256
#define LAT_  512
#define LD_   64

// ---------------------------------------------------------------------------
// async global->LDS 16-byte copy (CDNA5 GLOBAL_LOAD_ASYNC_TO_LDS_B128), with
// portable fallback. Waits are issued via s_wait_asynccnt before barriers.
#if defined(__has_builtin)
#if __has_builtin(__builtin_amdgcn_global_load_async_to_lds_b128)
#define HAVE_ASYNC_LDS 1
#endif
#endif

__device__ __forceinline__ void async_copy16(const _Float16* g, _Float16* l) {
#ifdef HAVE_ASYNC_LDS
    __builtin_amdgcn_global_load_async_to_lds_b128((v4i_*)g, (v4i_*)l, 0, 0);
#else
    *(uint4*)l = *(const uint4*)g;
#endif
}
__device__ __forceinline__ void async_wait_all() {
#ifdef HAVE_ASYNC_LDS
    asm volatile("s_wait_asynccnt 0x0" ::: "memory");
#endif
}

// ---------------------------------------------------------------------------
// plain f32 -> f16 cast (for ps_W)
__global__ void cast_f16_kernel(const float* __restrict__ in, _Float16* __restrict__ out, int n) {
    for (int i = blockIdx.x * blockDim.x + threadIdx.x; i < n; i += gridDim.x * blockDim.x)
        out[i] = (_Float16)in[i];
}

// conv weights: [COUT][CIN][3] f32 -> t-major f16: wr[co][t*CIN + ci]
__global__ void repack_w_kernel(const float* __restrict__ w, _Float16* __restrict__ wr,
                                int COUT, int CIN) {
    int KT = 3 * CIN, total = COUT * KT;
    for (int i = blockIdx.x * blockDim.x + threadIdx.x; i < total; i += gridDim.x * blockDim.x) {
        int co = i / KT, r = i % KT, t = r / CIN, ci = r % CIN;
        wr[i] = (_Float16)w[((size_t)co * CIN + ci) * 3 + t];
    }
}

// ---------------------------------------------------------------------------
// conv1: x[64,1,2048] f32 -> h1 channel-minor f16 [64][2048][64], leaky relu
__global__ void conv1_kernel(const float* __restrict__ x, const float* __restrict__ W1,
                             const float* __restrict__ b1, _Float16* __restrict__ h1) {
    int idx = blockIdx.x * blockDim.x + threadIdx.x;
    if (idx >= N_B * C1_ * L_IN) return;
    int co = idx % C1_;
    int l  = (idx / C1_) % L_IN;
    int n  = idx / (C1_ * L_IN);
    const float* xr = x + (size_t)n * L_IN;
    float s = b1[co];
    if (l > 0)        s += W1[co*3+0] * xr[l-1];
    s += W1[co*3+1] * xr[l];
    if (l < L_IN - 1) s += W1[co*3+2] * xr[l+1];
    s = s > 0.f ? s : 0.01f * s;
    h1[idx] = (_Float16)s;   // idx == (n*L_IN + l)*C1_ + co
}

// ---------------------------------------------------------------------------
// Implicit-GEMM conv1d (kernel=3,'same'), WMMA f32<-f16, t-major K ordering.
// in : [rows][LLEN][CIN] f16 (channel-minor), w: [COUT][3*CIN] f16 (t-major)
// out: channel-minor [rows][LLEN][COUT] (CMAJOR_OUT=false) or
//      channel-major [rows][COUT][LLEN] (CMAJOR_OUT=true, for final consumer)
template<int CIN, int COUT, int LLEN, int SPT, bool LEAKY, bool CMAJOR_OUT>
__global__ __launch_bounds__((SPT/16)*32)
void conv_wmma_kernel(const _Float16* __restrict__ in, const _Float16* __restrict__ w,
                      const float* __restrict__ bias, _Float16* __restrict__ out) {
    constexpr int KT = 3 * CIN;
    constexpr int NT = (SPT/16) * 32;
    constexpr int CH = CIN / 8;                      // 16B chunks per act row
    __shared__ alignas(32) _Float16 actT[SPT + 2][CIN];   // position-major tile + halo
    __shared__ alignas(32) _Float16 wS[16][KT];

    const int groupsPerRow = LLEN / SPT;
    int row = blockIdx.x / groupsPerRow;
    int l0  = (blockIdx.x % groupsPerRow) * SPT;
    int co0 = blockIdx.y * 16;
    int tid = threadIdx.x;

    // stage weights (contiguous rows of wRe) via async b128
    const _Float16* wr = w + (size_t)co0 * KT;
    for (int i = tid; i < 16 * (KT/8); i += NT) {
        int m = i / (KT/8), k8 = (i % (KT/8)) * 8;
        async_copy16(wr + (size_t)m * KT + k8, &wS[m][k8]);
    }
    // stage activation tile (+halo) channel-minor, zero boundary rows
    const _Float16* inr = in + (size_t)row * LLEN * CIN;
    for (int i = tid; i < (SPT + 2) * CH; i += NT) {
        int p = i / CH, c8 = (i % CH) * 8;
        int l = l0 + p - 1;
        if (l >= 0 && l < LLEN)
            async_copy16(inr + (size_t)l * CIN + c8, &actT[p][c8]);
        else
            *(uint4*)&actT[p][c8] = make_uint4(0u, 0u, 0u, 0u);
    }
    async_wait_all();
    __syncthreads();

    int wave = tid >> 5, lane = tid & 31, laneHi = lane >> 4, lm = lane & 15;
    int n0 = wave * 16;

    v8f acc = {};
#pragma unroll
    for (int kk = 0; kk < KT; kk += 32) {
        const int t   = kk / CIN;      // constant per step (CIN % 32 == 0)
        const int ci0 = kk % CIN;
        // A fragment: two contiguous 8-half runs
        v8h alo = *(const v8h*)&wS[lm][kk + laneHi * 16];
        v8h ahi = *(const v8h*)&wS[lm][kk + laneHi * 16 + 8];
        v16h a;
#pragma unroll
        for (int e = 0; e < 8; e++) { a[e] = alo[e]; a[e + 8] = ahi[e]; }
        // B fragment: one contiguous 16-half run (klB = e + laneHi*16)
        v16h b = *(const v16h*)&actT[n0 + lm + t][ci0 + laneHi * 16];
        acc = __builtin_amdgcn_wmma_f32_16x16x32_f16(false, a, false, b, (short)0, acc, false, false);
    }

    int pos = l0 + n0 + lm;
    if (CMAJOR_OUT) {
        _Float16* outr = out + (size_t)row * COUT * LLEN;
#pragma unroll
        for (int r = 0; r < 8; r++) {
            int co = co0 + r + laneHi * 8;
            float vv = acc[r] + bias[co];
            vv = LEAKY ? (vv > 0.f ? vv : 0.01f * vv) : (vv > 0.f ? vv : 0.f);
            outr[(size_t)co * LLEN + pos] = (_Float16)vv;
        }
    } else {
        // 8 consecutive channels per lane -> one b128 store
        _Float16* outr = out + ((size_t)row * LLEN + pos) * COUT + co0 + laneHi * 8;
        v8h ov;
#pragma unroll
        for (int r = 0; r < 8; r++) {
            float vv = acc[r] + bias[co0 + r + laneHi * 8];
            vv = LEAKY ? (vv > 0.f ? vv : 0.01f * vv) : (vv > 0.f ? vv : 0.f);
            ov[r] = (_Float16)vv;
        }
        *(v8h*)outr = ov;
    }
}

// ---------------------------------------------------------------------------
// Heads over channel-minor h3 [N][L][C3]: block = batch row, thread = channel.
// cm/sm/sl: [N,512], j = c*2 + d
__global__ __launch_bounds__(C3_)
void heads_kernel(const _Float16* __restrict__ h3,
                  const float* __restrict__ cmW, const float* __restrict__ cmb,
                  const float* __restrict__ smW, const float* __restrict__ smb,
                  const float* __restrict__ slW, const float* __restrict__ slb,
                  float* __restrict__ cm, float* __restrict__ sm, float* __restrict__ sl) {
    int n = blockIdx.x, c = threadIdx.x;
    const _Float16* hp = h3 + (size_t)n * L_IN * C3_ + c;
    float s0 = 0.f, s1 = 0.f, s2 = 0.f, s3 = 0.f, s4 = 0.f, s5 = 0.f;
    for (int l = 0; l < L_IN; l++) {
        float hv = (float)hp[(size_t)l * C3_];       // coalesced across threads
        float a0 = cmW[l], a1 = cmW[L_IN + l];        // uniform -> scalar loads
        float b0 = smW[l], b1 = smW[L_IN + l];
        float c0 = slW[l], c1 = slW[L_IN + l];
        s0 += hv * a0; s1 += hv * a1;
        s2 += hv * b0; s3 += hv * b1;
        s4 += hv * c0; s5 += hv * c1;
    }
    size_t base = (size_t)n * LAT_ + c * 2;
    cm[base] = s0 + cmb[0];  cm[base + 1] = s1 + cmb[1];
    sm[base] = s2 + smb[0];  sm[base + 1] = s3 + smb[1];
    sl[base] = s4 + slb[0];  sl[base + 1] = s5 + slb[1];
}

// group evidence fusion + content reparameterize (class_logvar == class_mu)
__global__ void zc_kernel(const float* __restrict__ cm, const float* __restrict__ eps_c,
                          float* __restrict__ z_c) {
    int j = threadIdx.x;
    float sp = 0.f, smp = 0.f;
    for (int n = 0; n < N_B; n++) {
        float mu = cm[(size_t)n * LAT_ + j];
        float p = __expf(-mu);
        sp += p; smp += mu * p;
    }
    float gv = 1.f / sp;
    z_c[j] = eps_c[j] * sqrtf(gv) + gv * smp;
}

__global__ void zs_kernel(const float* __restrict__ sm, const float* __restrict__ sl,
                          const float* __restrict__ eps_s, _Float16* __restrict__ z_s_h) {
    int i = blockIdx.x * blockDim.x + threadIdx.x;
    if (i >= N_B * LAT_) return;
    z_s_h[i] = (_Float16)(eps_s[i] * __expf(0.5f * sl[i]) + sm[i]);
}

// ---------------------------------------------------------------------------
// content projection GEMV -> decIn row 0, channel-minor [LD][C3]
__global__ __launch_bounds__(128)
void content_proj_kernel(const float* __restrict__ pcW, const float* __restrict__ pcb,
                         const float* __restrict__ z_c, _Float16* __restrict__ decIn) {
    __shared__ float zs[LAT_];
    for (int i = threadIdx.x; i < LAT_; i += 128) zs[i] = z_c[i];
    __syncthreads();
    int o = blockIdx.x * 128 + threadIdx.x;          // o = c*LD + l
    if (o >= C3_ * LD_) return;
    const float* wr2 = pcW + (size_t)o * LAT_;
    float s = pcb[o];
    for (int k = 0; k < LAT_; k++) s += wr2[k] * zs[k];
    decIn[(size_t)(o % LD_) * C3_ + (o / LD_)] = (_Float16)s;
}

// ---------------------------------------------------------------------------
// style projection GEMM (WMMA): out[n,o] = sum_k z_s[n,k]*ps_W[o,k] + ps_b[o]
// written channel-minor into decIn rows 1..64
__global__ __launch_bounds__(256)
void style_gemm_kernel(const _Float16* __restrict__ A,   // z_s  [64,512]
                       const _Float16* __restrict__ B,   // ps_W [16384,512] f16
                       const float* __restrict__ bias,   // ps_b
                       _Float16* __restrict__ outBase) { // decIn + C3*LD
    __shared__ alignas(32) _Float16 aS[16][LAT_];
    __shared__ alignas(32) _Float16 bS[128][32];
    int m0 = blockIdx.y * 16, col0 = blockIdx.x * 128;
    int tid = threadIdx.x;
    for (int i = tid; i < 16 * (LAT_/8); i += 256) {
        int m = i / (LAT_/8), k8 = (i % (LAT_/8)) * 8;
        async_copy16(A + (size_t)(m0 + m) * LAT_ + k8, &aS[m][k8]);
    }
    int wave = tid >> 5, lane = tid & 31, laneHi = lane >> 4, lm = lane & 15;
    v8f acc = {};
#pragma unroll 1
    for (int kk = 0; kk < LAT_; kk += 32) {
        __syncthreads();
        for (int i = tid; i < 128 * 4; i += 256) {
            int cc = i / 4, k8 = (i % 4) * 8;
            async_copy16(B + (size_t)(col0 + cc) * LAT_ + kk + k8, &bS[cc][k8]);
        }
        async_wait_all();
        __syncthreads();
        v8h alo = *(const v8h*)&aS[lm][kk + laneHi * 16];
        v8h ahi = *(const v8h*)&aS[lm][kk + laneHi * 16 + 8];
        v16h a;
#pragma unroll
        for (int e = 0; e < 8; e++) { a[e] = alo[e]; a[e + 8] = ahi[e]; }
        v16h b = *(const v16h*)&bS[wave * 16 + lm][laneHi * 16];
        acc = __builtin_amdgcn_wmma_f32_16x16x32_f16(false, a, false, b, (short)0, acc, false, false);
    }
    int col = col0 + wave * 16 + lm;                 // o = c*LD + l
    size_t pos = (size_t)(col % LD_) * C3_ + (col / LD_);
    float bcol = bias[col];
#pragma unroll
    for (int r = 0; r < 8; r++) {
        int n = m0 + r + laneHi * 8;
        outBase[(size_t)n * (C3_ * LD_) + pos] = (_Float16)(acc[r] + bcol);
    }
}

// ---------------------------------------------------------------------------
// full 1D convolution with LDS-staged operands:
// out[n][j] = sum_i media[n][i]*source[j-i], both length 4096 (channel-major flat)
__global__ __launch_bounds__(256)
void fullconv_kernel(const _Float16* __restrict__ dec,  // [65][4096], row0 = source
                     float* __restrict__ out) {
    const int ML = C1_ * LD_;                 // 4096
    __shared__ _Float16 srcS[ML];
    __shared__ _Float16 mS[ML];
    int n = blockIdx.y;
    const _Float16* mrow = dec + (size_t)(1 + n) * ML;
    for (int i = threadIdx.x; i < ML / 8; i += 256) {
        *(v8h*)&srcS[i * 8] = *(const v8h*)&dec[i * 8];
        *(v8h*)&mS[i * 8]   = *(const v8h*)&mrow[i * 8];
    }
    __syncthreads();
    int j = blockIdx.x * 256 + threadIdx.x;
    if (j >= 2 * ML - 1) return;
    int ilo = j - (ML - 1); if (ilo < 0) ilo = 0;
    int ihi = j < ML - 1 ? j : ML - 1;
    float s = 0.f;
    for (int i = ilo; i <= ihi; i++) s += (float)mS[i] * (float)srcS[j - i];
    out[(size_t)n * (2 * ML - 1) + j] = s;
}

// ---------------------------------------------------------------------------
extern "C" void kernel_launch(void* const* d_in, const int* in_sizes, int n_in,
                              void* d_out, int out_size, void* d_ws, size_t ws_size,
                              hipStream_t stream) {
    (void)in_sizes; (void)n_in; (void)out_size; (void)ws_size;
    const float* x    = (const float*)d_in[0];
    const float* W1   = (const float*)d_in[1];  const float* b1 = (const float*)d_in[2];
    const float* W2   = (const float*)d_in[3];  const float* b2 = (const float*)d_in[4];
    const float* W3   = (const float*)d_in[5];  const float* b3 = (const float*)d_in[6];
    const float* cmW  = (const float*)d_in[7];  const float* cmb = (const float*)d_in[8];
    const float* smW  = (const float*)d_in[9];  const float* smb = (const float*)d_in[10];
    const float* slW  = (const float*)d_in[11]; const float* slb = (const float*)d_in[12];
    const float* pcW  = (const float*)d_in[13]; const float* pcb = (const float*)d_in[14];
    const float* psW  = (const float*)d_in[15]; const float* psb = (const float*)d_in[16];
    const float* V1   = (const float*)d_in[17]; const float* c1 = (const float*)d_in[18];
    const float* V2   = (const float*)d_in[19]; const float* c2 = (const float*)d_in[20];
    const float* epsC = (const float*)d_in[21]; const float* epsS = (const float*)d_in[22];
    float* out = (float*)d_out;

    char* wsp = (char*)d_ws;
    size_t off = 0;
    auto alloc = [&](size_t bytes) -> void* {
        void* p = wsp + off;
        off = (off + bytes + 255) & ~(size_t)255;
        return p;
    };
    _Float16* h1    = (_Float16*)alloc((size_t)N_B * C1_ * L_IN * 2);   // [n][l][c]
    _Float16* h2    = (_Float16*)alloc((size_t)N_B * C2_ * L_IN * 2);   // [n][l][c]
    _Float16* h3    = (_Float16*)alloc((size_t)N_B * C3_ * L_IN * 2);   // [n][l][c]
    _Float16* W2h   = (_Float16*)alloc((size_t)C2_ * C1_ * 3 * 2);      // t-major
    _Float16* W3h   = (_Float16*)alloc((size_t)C3_ * C2_ * 3 * 2);
    _Float16* V1h   = (_Float16*)alloc((size_t)C2_ * C3_ * 3 * 2);
    _Float16* V2h   = (_Float16*)alloc((size_t)C1_ * C2_ * 3 * 2);
    _Float16* psWh  = (_Float16*)alloc((size_t)C3_ * LD_ * LAT_ * 2);
    float*    cm    = (float*)alloc((size_t)N_B * LAT_ * 4);
    float*    sm    = (float*)alloc((size_t)N_B * LAT_ * 4);
    float*    sl    = (float*)alloc((size_t)N_B * LAT_ * 4);
    float*    z_c   = (float*)alloc(LAT_ * 4);
    _Float16* z_s_h = (_Float16*)alloc((size_t)N_B * LAT_ * 2);
    _Float16* decIn  = (_Float16*)alloc((size_t)(N_B + 1) * C3_ * LD_ * 2);  // [row][l][c]
    _Float16* decMid = (_Float16*)alloc((size_t)(N_B + 1) * C2_ * LD_ * 2);  // [row][l][c]
    _Float16* decOut = (_Float16*)alloc((size_t)(N_B + 1) * C1_ * LD_ * 2);  // [row][c][l]

    // 0) weight repack/cast
    repack_w_kernel<<<64, 256, 0, stream>>>(W2, W2h, C2_, C1_);
    repack_w_kernel<<<128, 256, 0, stream>>>(W3, W3h, C3_, C2_);
    repack_w_kernel<<<128, 256, 0, stream>>>(V1, V1h, C2_, C3_);
    repack_w_kernel<<<64, 256, 0, stream>>>(V2, V2h, C1_, C2_);
    cast_f16_kernel<<<1024, 256, 0, stream>>>(psW, psWh, C3_ * LD_ * LAT_);

    // 1) encoder
    {
        int total = N_B * C1_ * L_IN;
        conv1_kernel<<<(total + 255) / 256, 256, 0, stream>>>(x, W1, b1, h1);
    }
    conv_wmma_kernel<C1_, C2_, L_IN, 128, true, false>
        <<<dim3(N_B * (L_IN / 128), C2_ / 16), 256, 0, stream>>>(h1, W2h, b2, h2);
    conv_wmma_kernel<C2_, C3_, L_IN, 128, true, false>
        <<<dim3(N_B * (L_IN / 128), C3_ / 16), 256, 0, stream>>>(h2, W3h, b3, h3);

    // 2) heads + fusion + reparameterize
    heads_kernel<<<N_B, C3_, 0, stream>>>(h3, cmW, cmb, smW, smb, slW, slb, cm, sm, sl);
    zc_kernel<<<1, LAT_, 0, stream>>>(cm, epsC, z_c);
    zs_kernel<<<(N_B * LAT_ + 255) / 256, 256, 0, stream>>>(sm, sl, epsS, z_s_h);

    // 3) projections -> decoder input (row 0 = content, rows 1..64 = media)
    content_proj_kernel<<<(C3_ * LD_) / 128, 128, 0, stream>>>(pcW, pcb, z_c, decIn);
    style_gemm_kernel<<<dim3((C3_ * LD_) / 128, N_B / 16), 256, 0, stream>>>(
        z_s_h, psWh, psb, decIn + (size_t)C3_ * LD_);

    // 4) decoder convs over all 65 rows (relu); last one channel-major out
    conv_wmma_kernel<C3_, C2_, LD_, 64, false, false>
        <<<dim3(N_B + 1, C2_ / 16), 128, 0, stream>>>(decIn, V1h, c1, decMid);
    conv_wmma_kernel<C2_, C1_, LD_, 64, false, true>
        <<<dim3(N_B + 1, C1_ / 16), 128, 0, stream>>>(decMid, V2h, c2, decOut);

    // 5) final full 1D convolution (f32, LDS-staged)
    {
        int outLen = 2 * C1_ * LD_ - 1;  // 8191
        fullconv_kernel<<<dim3((outLen + 255) / 256, N_B), 256, 0, stream>>>(decOut, out);
    }
}
